// Reference_9079560864086
// MI455X (gfx1250) — matrix operand order repaired and checked by reading, not compiled, not run on hardware
//
#include <hip/hip_runtime.h>
#include <cstdint>
#include <cstddef>

typedef __attribute__((ext_vector_type(16))) _Float16 v16h;
typedef __attribute__((ext_vector_type(8)))  float    v8f;

#define D_IN   128
#define D_SH   16
#define D_OUT  128
#define NPATH  64
#define TILE_E 16
#define WAVES_PER_BLOCK 2
#define BLOCK_THREADS   (WAVES_PER_BLOCK * 32)
#define STAGE_LOADS_HEX "0x12"   // 18 async loads per stage (16 x-rows + 2 sh)

// Low 32 bits of a flat shared-aperture address == LDS byte address.
__device__ __forceinline__ uint32_t lds_addr_of(const void* p) {
    return (uint32_t)(uintptr_t)p;
}

__global__ __launch_bounds__(BLOCK_THREADS)
void edge_tp_wmma_kernel(const float* __restrict__ x,
                         const float* __restrict__ sh,
                         const float* __restrict__ coeff,
                         const long long* __restrict__ src,
                         const long long* __restrict__ dst,
                         const int* __restrict__ ki,
                         const int* __restrict__ kj,
                         const int* __restrict__ ko,
                         float* __restrict__ out,
                         int nEdges, int nTiles)
{
    // Double-buffered per-wave staging: 36 KB static LDS per 64-thread block.
    __shared__ __align__(16) float xbuf [WAVES_PER_BLOCK][2][TILE_E][D_IN];
    __shared__ __align__(16) float shbuf[WAVES_PER_BLOCK][2][TILE_E][D_SH];

    const int lane = threadIdx.x & 31;
    const int wave = threadIdx.x >> 5;
    const int hi   = lane >> 4;   // which half-wave
    const int lo   = lane & 15;   // row (A) / column (B,D) index

    // ---- Per-lane loop-invariant path metadata -------------------------
    // A-matrix K mapping (16-bit A 16x32, ISA 7.12.2):
    //   half h of v16h: K = (h<8 ? h : h+8) + 8*hi
    // B-matrix K mapping (same K order as A):
    //   half h of v16h: K = (h<8 ? h : h+8) + 8*hi, column N = lo
    int  kij[2][16];   // packed ki | (kj<<16) for the A build
    v16h B[2][8];      // B tiles: [K-half][output 16-col tile], coeff folded in
#pragma unroll
    for (int kh = 0; kh < 2; ++kh) {
#pragma unroll
        for (int h = 0; h < 16; ++h) {
            const int pA = 32 * kh + ((h < 8) ? h : h + 8) + 8 * hi;
            kij[kh][h] = ki[pA] | (kj[pA] << 16);

            const int   pB = 32 * kh + ((h < 8) ? h : h + 8) + 8 * hi;
            const int   o  = ko[pB];
            const float c  = coeff[pB];
#pragma unroll
            for (int t = 0; t < 8; ++t) {
                B[kh][t][h] = (o == t * 16 + lo) ? (_Float16)c : (_Float16)0.0f;
            }
        }
    }

    const int waveId     = blockIdx.x * WAVES_PER_BLOCK + wave;
    const int waveStride = gridDim.x * WAVES_PER_BLOCK;

    // Issue one stage (18 async b128 loads) into buffer `buf`.
    auto issueStage = [&](int tile, int buf) {
        const int e0 = tile * TILE_E;
        // gathered x rows: one b128/lane covers one 512B row
#pragma unroll 4
        for (int r = 0; r < TILE_E; ++r) {
            int e = e0 + r; if (e >= nEdges) e = nEdges - 1;
            const long long node  = src[e];
            const float*    gaddr = x + (size_t)node * D_IN + lane * 4;
            const uint32_t  laddr = lds_addr_of(&xbuf[wave][buf][r][lane * 4]);
            asm volatile("global_load_async_to_lds_b128 %0, %1, off"
                         :: "v"(laddr), "v"(gaddr) : "memory");
        }
        // sh rows: 16 edges x 16 f32 in 2 wave-wide b128s
#pragma unroll
        for (int q = 0; q < 2; ++q) {
            const int flat = q * 32 + lane;
            const int r    = flat >> 2;
            const int c0   = (flat & 3) * 4;
            int e = e0 + r; if (e >= nEdges) e = nEdges - 1;
            const float*   gaddr = sh + (size_t)e * D_SH + c0;
            const uint32_t laddr = lds_addr_of(&shbuf[wave][buf][r][c0]);
            asm volatile("global_load_async_to_lds_b128 %0, %1, off"
                         :: "v"(laddr), "v"(gaddr) : "memory");
        }
    };

    // ---- Software pipeline: prefetch next tile while computing current.
    int buf = 0;
    if (waveId < nTiles) issueStage(waveId, 0);

    for (int tile = waveId; tile < nTiles; tile += waveStride) {
        const int e0       = tile * TILE_E;
        const int nextTile = tile + waveStride;

        if (nextTile < nTiles) {
            issueStage(nextTile, buf ^ 1);
            // ASYNCcnt completes in order: <=18 outstanding means the 18
            // loads of the *current* buffer (issued last iteration) are done.
            asm volatile("s_wait_asynccnt " STAGE_LOADS_HEX ::: "memory");
        } else {
            asm volatile("s_wait_asynccnt 0x0" ::: "memory");
        }

        // ---- build A (vals = xg * sh) in f16, two K-halves -------------
        v16h A0, A1;
#pragma unroll
        for (int h = 0; h < 16; ++h) {
            {
                const int idx = kij[0][h];
                const float xa = xbuf [wave][buf][lo][idx & 0xffff];
                const float sv = shbuf[wave][buf][lo][idx >> 16];
                A0[h] = (_Float16)(xa * sv);
            }
            {
                const int idx = kij[1][h];
                const float xa = xbuf [wave][buf][lo][idx & 0xffff];
                const float sv = shbuf[wave][buf][lo][idx >> 16];
                A1[h] = (_Float16)(xa * sv);
            }
        }

        // ---- msg[16x128] = A(16x64) @ B(64x128): 16 WMMAs --------------
        v8f acc[8];
#pragma unroll
        for (int t = 0; t < 8; ++t) {
            v8f c = {};
            c = __builtin_amdgcn_wmma_f32_16x16x32_f16(false, A0, false, B[0][t],
                                                       (short)0, c, false, false);
            c = __builtin_amdgcn_wmma_f32_16x16x32_f16(false, A1, false, B[1][t],
                                                       (short)0, c, false, false);
            acc[t] = c;
        }

        // ---- scatter-add into out[dst[e]] ------------------------------
        // D layout: lane holds rows M = r + 8*hi (r = acc VGPR idx), col = lo.
#pragma unroll
        for (int r = 0; r < 8; ++r) {
            const int e = e0 + r + 8 * hi;
            if (e < nEdges) {
                const long long node = dst[e];
                const uint32_t  base =
                    (uint32_t)(((uint32_t)node * D_OUT + lo) * sizeof(float));
#pragma unroll
                for (int t = 0; t < 8; ++t) {
                    const uint32_t off = base + (uint32_t)(t * 16 * sizeof(float));
                    const float    v   = acc[t][r];
                    asm volatile("global_atomic_add_f32 %0, %1, %2 scope:SCOPE_DEV"
                                 :: "v"(off), "v"(v), "s"(out) : "memory");
                }
            }
        }

        buf ^= 1;
    }
}

extern "C" void kernel_launch(void* const* d_in, const int* in_sizes, int n_in,
                              void* d_out, int out_size, void* d_ws, size_t ws_size,
                              hipStream_t stream) {
    const float*     x     = (const float*)d_in[0];
    const float*     sh    = (const float*)d_in[1];
    const float*     coeff = (const float*)d_in[2];
    const long long* src   = (const long long*)d_in[3];
    const long long* dst   = (const long long*)d_in[4];
    const int*       ki    = (const int*)d_in[5];
    const int*       kj    = (const int*)d_in[6];
    const int*       ko    = (const int*)d_in[7];
    float*           out   = (float*)d_out;

    const int nEdges = in_sizes[3];
    const int nTiles = (nEdges + TILE_E - 1) / TILE_E;

    // out is accumulated with atomics: zero it first (graph-capturable).
    hipMemsetAsync(d_out, 0, (size_t)out_size * sizeof(float), stream);

    int blocks = (nTiles + WAVES_PER_BLOCK - 1) / WAVES_PER_BLOCK;
    if (blocks > 2048) blocks = 2048;   // ~12 tiles/wave amortizes B-matrix setup
    edge_tp_wmma_kernel<<<blocks, BLOCK_THREADS, 0, stream>>>(
        x, sh, coeff, src, dst, ki, kj, ko, out, nEdges, nTiles);
}
